// AttLIF_40973988004674
// MI455X (gfx1250) — matrix operand order repaired and checked by reading, not compiled, not run on hardware
//
#include <hip/hip_runtime.h>
#include <hip/hip_bf16.h>
#include <math.h>

// Problem constants (match reference)
#define BATCH 256
#define TSTEPS 64
#define DIN 1024
#define DH 2048
#define MROWS (BATCH * TSTEPS)   // 16384 GEMM rows
#define VTH 0.6f

typedef __attribute__((ext_vector_type(16))) __bf16 v16bf;
typedef __attribute__((ext_vector_type(8)))  float  v8f;

// -------- split-precision conversion: a = hi + lo, hi = top-16-bit truncation
// (exact bit extraction), lo = RNE-rounded residual. Dropping lo*lo in the
// product leaves ~2^-17 relative error -> effectively fp32-accurate GEMM on
// the bf16 WMMA pipe.
__device__ __forceinline__ void cvt8_split(v16bf& fh, v16bf& fl, int base,
                                           const float* __restrict__ p) {
#pragma unroll
    for (int e = 0; e < 8; ++e) {
        const float a  = p[e];
        const float ah = __uint_as_float(__float_as_uint(a) & 0xFFFF0000u);
        fh[base + e] = (__bf16)ah;        // exact (low mantissa already zero)
        fl[base + e] = (__bf16)(a - ah);  // residual, RNE
    }
}

// =====================================================================
// Kernel 1: x[row, h] = sum_k data[row,k] * W[h,k] + bias[h]
// 256 threads = 8 waves; wave tile 64x32 (4x2 of 16x16); block tile 128x128.
// A fragment (16x32 bf16, same K order as B): lane m=L&15, kg=L>>4 holds
//   elems 0..7  -> K = kg*16 + e
//   elems 8..15 -> K = kg*16 + e
// B fragment (32x16 bf16): lane n=L&15, half=L>>4 holds elems e -> K = half*16 + e
// Both map to contiguous per-lane loads because data and W are K-contiguous.
// Per tile: acc += A_lo*B_hi + A_hi*B_lo + A_hi*B_hi (smallest terms first).
// =====================================================================
__global__ __launch_bounds__(256) void attlif_gemm(
    const float* __restrict__ A,     // [MROWS, DIN]
    const float* __restrict__ Wm,    // [DH, DIN]
    const float* __restrict__ bias,  // [DH]
    float* __restrict__ C)           // [MROWS, DH]
{
    const int lane  = threadIdx.x & 31;
    const int wave  = threadIdx.x >> 5;
    const int waveM = wave >> 2;               // 0..1
    const int waveN = wave & 3;                // 0..3
    const int rowBase = blockIdx.y * 128 + waveM * 64;
    const int colBase = blockIdx.x * 128 + waveN * 32;

    const int ml = lane & 15;                  // M (A) or N (B) index within tile
    const int kg = lane >> 4;                  // K-group select

    v8f acc[4][2];
#pragma unroll
    for (int i = 0; i < 4; ++i)
#pragma unroll
        for (int j = 0; j < 2; ++j)
            acc[i][j] = (v8f){0.f, 0.f, 0.f, 0.f, 0.f, 0.f, 0.f, 0.f};

    for (int k0 = 0; k0 < DIN; k0 += 32) {
        v16bf ah[4], al[4];
#pragma unroll
        for (int i = 0; i < 4; ++i) {
            const float* ap = A + (size_t)(rowBase + i * 16 + ml) * DIN + k0;
            cvt8_split(ah[i], al[i], 0, ap + kg * 16);      // K = k0 + kg*16 + 0..7
            cvt8_split(ah[i], al[i], 8, ap + kg * 16 + 8);  // K = k0 + kg*16 + 8..15
        }
        v16bf bh[2], bl[2];
#pragma unroll
        for (int j = 0; j < 2; ++j) {
            const float* bp = Wm + (size_t)(colBase + j * 16 + ml) * DIN + k0 + kg * 16;
            cvt8_split(bh[j], bl[j], 0, bp);                // K = k0 + kg*16 + 0..7
            cvt8_split(bh[j], bl[j], 8, bp + 8);            // K = k0 + kg*16 + 8..15
        }
#pragma unroll
        for (int i = 0; i < 4; ++i)
#pragma unroll
            for (int j = 0; j < 2; ++j) {
                acc[i][j] = __builtin_amdgcn_wmma_f32_16x16x32_bf16(
                    false, al[i], false, bh[j], (short)0, acc[i][j], false, false);
                acc[i][j] = __builtin_amdgcn_wmma_f32_16x16x32_bf16(
                    false, ah[i], false, bl[j], (short)0, acc[i][j], false, false);
                acc[i][j] = __builtin_amdgcn_wmma_f32_16x16x32_bf16(
                    false, ah[i], false, bh[j], (short)0, acc[i][j], false, false);
            }
    }

    // Epilogue: C VGPR r holds M = r (lanes 0-15) / r+8 (lanes 16-31), N = lane&15
#pragma unroll
    for (int i = 0; i < 4; ++i) {
#pragma unroll
        for (int j = 0; j < 2; ++j) {
            const int col = colBase + j * 16 + ml;
            const float bv = bias[col];
#pragma unroll
            for (int r = 0; r < 8; ++r) {
                const int row = rowBase + i * 16 + r + 8 * kg;
                C[(size_t)row * DH + col] = acc[i][j][r] + bv;
            }
        }
    }
}

// =====================================================================
// Kernel 2: per-row mean and max over DH (row = b*T + t), 16384 blocks.
// =====================================================================
__global__ __launch_bounds__(256) void attlif_reduce(
    const float* __restrict__ x, float* __restrict__ avg, float* __restrict__ mx)
{
    const int row = blockIdx.x;
    const int tid = threadIdx.x;
    const float* p = x + (size_t)row * DH;

    float s = 0.f, m = -INFINITY;
    for (int i = tid; i < DH; i += 256) {
        const float v = p[i];
        s += v;
        m = fmaxf(m, v);
    }
    __shared__ float ssum[256], smax[256];
    ssum[tid] = s; smax[tid] = m;
    __syncthreads();
    for (int off = 128; off > 0; off >>= 1) {
        if (tid < off) {
            ssum[tid] += ssum[tid + off];
            smax[tid] = fmaxf(smax[tid], smax[tid + off]);
        }
        __syncthreads();
    }
    if (tid == 0) {
        avg[row] = ssum[0] * (1.f / (float)DH);
        mx[row]  = smax[0];
    }
}

// =====================================================================
// Kernel 3: score = sigmoid( W2@relu(W1@avg) + W2@relu(W1@mx) ), per batch b.
// =====================================================================
__global__ __launch_bounds__(64) void attlif_score(
    const float* __restrict__ avg, const float* __restrict__ mx,
    const float* __restrict__ W1,  // [4, 64]
    const float* __restrict__ W2,  // [64, 4]
    float* __restrict__ score)     // [B, T]
{
    __shared__ float sa[TSTEPS], sm[TSTEPS];
    const int b = blockIdx.x;
    const int t = threadIdx.x;
    sa[t] = avg[b * TSTEPS + t];
    sm[t] = mx[b * TSTEPS + t];
    __syncthreads();

    float acc = 0.f;
#pragma unroll
    for (int r = 0; r < 4; ++r) {
        float ha = 0.f, hm = 0.f;
#pragma unroll 8
        for (int tt = 0; tt < TSTEPS; ++tt) {
            const float w = W1[r * TSTEPS + tt];
            ha += sa[tt] * w;
            hm += sm[tt] * w;
        }
        const float w2 = W2[t * 4 + r];
        acc += (fmaxf(ha, 0.f) + fmaxf(hm, 0.f)) * w2;
    }
    score[b * TSTEPS + t] = 1.f / (1.f + __expf(-acc));
}

// =====================================================================
// Kernel 4: IF neuron scan over T, in place over x (=d_out).
// Thread owns column (b, h); reads x[b,t,h], writes spike back.
// =====================================================================
__global__ __launch_bounds__(256) void attlif_scan(
    float* __restrict__ x, const float* __restrict__ score)
{
    __shared__ float sc[TSTEPS];
    const int b = blockIdx.y;
    const int h = blockIdx.x * 256 + threadIdx.x;
    if (threadIdx.x < TSTEPS) sc[threadIdx.x] = score[b * TSTEPS + threadIdx.x];
    __syncthreads();

    float v = 0.f;
    const size_t base = (size_t)b * TSTEPS * DH + h;
#pragma unroll 4
    for (int t = 0; t < TSTEPS; ++t) {
        const size_t idx = base + (size_t)t * DH;
        v += x[idx] * sc[t];
        const float s = (v - VTH >= 0.f) ? 1.f : 0.f;
        x[idx] = s;
        v = (s != 0.f) ? 0.f : v;   // hard reset to 0
    }
}

// =====================================================================
extern "C" void kernel_launch(void* const* d_in, const int* in_sizes, int n_in,
                              void* d_out, int out_size, void* d_ws, size_t ws_size,
                              hipStream_t stream) {
    (void)in_sizes; (void)n_in; (void)out_size; (void)ws_size;
    const float* data = (const float*)d_in[0];   // [B,T,DIN]
    const float* Wm   = (const float*)d_in[1];   // [DH,DIN]
    const float* bias = (const float*)d_in[2];   // [DH]
    const float* W1   = (const float*)d_in[3];   // [4,64]
    const float* W2   = (const float*)d_in[4];   // [64,4]
    float* out = (float*)d_out;                  // [B,T,DH] — reused as x buffer

    float* avg   = (float*)d_ws;                 // [B*T]
    float* mx    = avg + MROWS;                  // [B*T]
    float* score = mx + MROWS;                   // [B*T]

    // 1) GEMM: x = data @ W^T + bias  -> d_out (bf16x3 split, f32 accumulate)
    dim3 gGemm(DH / 128, MROWS / 128);           // (16, 128)
    attlif_gemm<<<gGemm, 256, 0, stream>>>(data, Wm, bias, out);

    // 2) per-(b,t) mean/max over DH
    attlif_reduce<<<MROWS, 256, 0, stream>>>(out, avg, mx);

    // 3) temporal-attention gate
    attlif_score<<<BATCH, TSTEPS, 0, stream>>>(avg, mx, W1, W2, score);

    // 4) gated IF scan, in place over d_out
    dim3 gScan(DH / 256, BATCH);                 // (8, 256)
    attlif_scan<<<gScan, 256, 0, stream>>>(out, score);
}
